// LSTMForecaster_8315056685557
// MI455X (gfx1250) — matrix operand order repaired and checked by reading, not compiled, not run on hardware
//
#include <hip/hip_runtime.h>
#include <hip/hip_bf16.h>
#include <stddef.h>

// ---------------------------------------------------------------------------
// LSTM forecaster on MI455X (gfx1250, wave32, WMMA + TDM).
//
// Per-layer recurrent kernel keeps the whole packed bf16 weight matrix
// [W_ih | W_hh] resident in LDS (<=264KB of the 320KB WGP LDS), loaded once
// via the Tensor Data Mover (tensor_load_to_lds). 8 workgroups (one per
// 16-row batch tile) iterate 512 timesteps computing
//   gates = [x_t ; h] @ W^T + bias   with v_wmma_f32_16x16x32_bf16.
// Input columns are double-buffered in LDS so next-step global loads overlap
// the current step's WMMA + element-wise work (2 barriers per step).
// Element-wise uses hardware V_TANH (sigmoid via 0.5*(1+tanh(x/2))).
// ---------------------------------------------------------------------------

#define BB 128
#define TT 512
#define DD 32
#define HH 128
#define GG 512   // 4*H
#define FF 4

typedef __attribute__((ext_vector_type(16))) __bf16       v16bf;
typedef __attribute__((ext_vector_type(8)))  float        v8f;
typedef __attribute__((ext_vector_type(4)))  unsigned int v4u;
typedef __attribute__((ext_vector_type(8)))  int          v8i;
typedef __attribute__((ext_vector_type(4)))  int          v4i;

#if __has_builtin(__builtin_amdgcn_tensor_load_to_lds) && \
    __has_builtin(__builtin_amdgcn_s_wait_tensorcnt)
#define USE_TDM 1
#else
#define USE_TDM 0
#endif

#if __has_builtin(__builtin_amdgcn_tanhf)
__device__ __forceinline__ float fast_tanh(float x) { return __builtin_amdgcn_tanhf(x); }
#elif __has_builtin(__builtin_amdgcn_tanh_f32)
__device__ __forceinline__ float fast_tanh(float x) { return __builtin_amdgcn_tanh_f32(x); }
#else
__device__ __forceinline__ float fast_tanh(float x) { return tanhf(x); }
#endif
__device__ __forceinline__ float fast_sigmoid(float x) {
    return 0.5f * (1.f + fast_tanh(0.5f * x));     // exact identity
}

__device__ __forceinline__ unsigned short f2bf(float x) {
    unsigned int u = __builtin_bit_cast(unsigned int, x);
    u += 0x7fffu + ((u >> 16) & 1u);               // round-to-nearest-even
    return (unsigned short)(u >> 16);
}
__device__ __forceinline__ unsigned short to_bf_in(float x)          { return f2bf(x); }
__device__ __forceinline__ unsigned short to_bf_in(unsigned short x) { return x; }
__device__ __forceinline__ void store_h(float* p, float h)          { *p = h; }
__device__ __forceinline__ void store_h(unsigned short* p, float h) { *p = f2bf(h); }

// Load a 32-byte WMMA fragment (16 bf16) from LDS as two b128 loads.
struct Frag2 { uint4 lo, hi; };
__device__ __forceinline__ v16bf load_frag2(const unsigned short* lo,
                                            const unsigned short* hi) {
    Frag2 f;
    f.lo = *reinterpret_cast<const uint4*>(lo);
    f.hi = *reinterpret_cast<const uint4*>(hi);
    return __builtin_bit_cast(v16bf, f);
}

// ---------------------------------------------------------------------------
// Weight packing: bf16 [W_ih | W_hh] per layer (row = gate output n, padded
// row stride), plus combined bias.
//   layer 0  : [512][168]  (32 x-cols + 128 h-cols + 8 pad)
//   layer 1,2: [512][264]  (128 + 128 + 8 pad)
// ---------------------------------------------------------------------------
__global__ void pack_weights_kernel(const float* __restrict__ w_ih0,
                                    const float* __restrict__ w_ih_rest,
                                    const float* __restrict__ w_hh,
                                    const float* __restrict__ b_ih,
                                    const float* __restrict__ b_hh,
                                    unsigned short* __restrict__ wp0,
                                    unsigned short* __restrict__ wp12,
                                    float* __restrict__ biases) {
    const int stride = gridDim.x * blockDim.x;
    const int tid0 = blockIdx.x * blockDim.x + threadIdx.x;

    for (int i = tid0; i < GG * 168; i += stride) {
        int n = i / 168, k = i % 168;
        float v = 0.f;
        if (k < DD)            v = w_ih0[n * DD + k];
        else if (k < DD + HH)  v = w_hh[n * HH + (k - DD)];     // layer 0
        wp0[i] = f2bf(v);
    }
    for (int i = tid0; i < 2 * GG * 264; i += stride) {
        int l = i / (GG * 264);
        int r = i % (GG * 264);
        int n = r / 264, k = r % 264;
        float v = 0.f;
        if (k < HH)            v = w_ih_rest[(l * GG + n) * HH + k];
        else if (k < 2 * HH)   v = w_hh[((l + 1) * GG + n) * HH + (k - HH)];
        wp12[i] = f2bf(v);
    }
    for (int i = tid0; i < 3 * GG; i += stride) biases[i] = b_ih[i] + b_hh[i];
}

// ---------------------------------------------------------------------------
// Recurrent LSTM layer. One workgroup per 16-row batch tile; 1024 threads =
// 32 waves; each wave owns one 16-wide slice of the 512 gate outputs.
// ---------------------------------------------------------------------------
template <int DIN, typename TIN, typename TOUT>
__global__ __launch_bounds__(1024, 1)
void lstm_layer_kernel(const TIN* __restrict__ xin,             // [B][T][DIN]
                       const unsigned short* __restrict__ wpack,// [512][WROW] bf16
                       const float* __restrict__ bias,          // [512]
                       TOUT* __restrict__ hout)                 // [B][T][H]
{
    constexpr int KD    = DIN + HH;   // GEMM K
    constexpr int WROW  = KD + 8;     // padded weight row stride
    constexpr int AXROW = DIN + 8;    // padded x-region row stride
    constexpr int AHROW = HH + 8;     // padded h-region row stride
    constexpr int GROW  = GG + 4;     // padded gates row stride
    constexpr int XC    = DIN / 32;   // x K-chunks (1 or 4)
    constexpr int HC    = HH / 32;    // h K-chunks (4)

    extern __shared__ char smem[];
    unsigned short* Wl = (unsigned short*)smem;        // [512][WROW]
    unsigned short* Ax = Wl + GG * WROW;               // [2][16][AXROW] x_t (dbl buf)
    unsigned short* Ah = Ax + 2 * 16 * AXROW;          // [16][AHROW]    h
    float*          Gl = (float*)(Ah + 16 * AHROW);    // [16][GROW] gate accum

    const int tid = threadIdx.x;
    const int b0  = blockIdx.x * 16;
    const int wav = tid >> 5;                  // 0..31 -> gate N-tile

    // ---- one-time: whole layer weight matrix -> LDS ----------------------
#if USE_TDM
    if (wav == 0) {
        // 1-D TDM copy: NE elements of 8 bytes, single tile.
        constexpr unsigned NE = (unsigned)((GG * WROW) / 4); // 8B elements (<65536)
        const unsigned lds_off = (unsigned)(size_t)(void*)Wl;   // low 32b of flat LDS addr
        const unsigned long long ga = (unsigned long long)(size_t)wpack;
        v4u g0 = { 1u,                                   // count=1, user mode
                   lds_off,                              // lds_addr
                   (unsigned)(ga & 0xffffffffu),         // global_addr[31:0]
                   (unsigned)((ga >> 32) & 0x01ffffffu)  // global_addr[56:32]
                       | 0x80000000u };                  // type=2 ("image")
        v8i g1 = { (int)0x00030000u,                     // wg_mask=0, data_size=8B
                   (int)((NE & 0xffffu) << 16),          // tensor_dim0[15:0]
                   (int)((NE >> 16) | (1u << 16)),       // tensor_dim0[31:16], tensor_dim1=1
                   (int)((NE & 0xffffu) << 16),          // tile_dim0 = NE
                   1,                                    // tile_dim1 = 1
                   (int)NE,                              // tensor_dim0_stride[31:0]
                   (int)((NE & 0xffffu) << 16),          // stride0 hi=0 | stride1[15:0]
                   0 };                                  // stride1[47:16]
        v4i z4 = { 0, 0, 0, 0 };
        v8i z8 = { 0, 0, 0, 0, 0, 0, 0, 0 };
        // 6-arg toolchain variant: (g0, g1, g2, g3, extra_group, cpol)
        __builtin_amdgcn_tensor_load_to_lds(g0, g1, z4, z4, z8, 0);
        __builtin_amdgcn_s_wait_tensorcnt(0);
    }
#else
    {
        const uint4* src = (const uint4*)wpack;
        uint4*       dst = (uint4*)Wl;
        for (int i = tid; i < (GG * WROW) / 8; i += 1024) dst[i] = src[i];
    }
#endif
    for (int i = tid; i < 16 * AHROW; i += 1024) Ah[i] = 0;   // h(-1) = 0
    // prologue: stage x_0 into buffer 0
    for (int i = tid; i < 16 * DIN; i += 1024) {
        int r = i / DIN, d = i % DIN;
        Ax[r * AXROW + d] = to_bf_in(xin[(((size_t)(b0 + r)) * TT + 0) * DIN + d]);
    }

    const int lane  = tid & 31;
    const int nl    = lane & 15;
    const int ncol  = wav * 16 + nl;           // gate column this lane produces
    const float bn  = bias[ncol];
    // A-frag layout (ISA 16-bit A 16x32): lanes 0-15 row M=lane, K {0..7,16..23};
    // lanes 16-31 row M=lane-16, K {8..15,24..31}.
    const int arow  = nl;
    const int koffA = (lane < 16) ? 0 : 8;
    // B-frag: lane holds one gate column; lanes 0-15 K {0..7,16..23}, 16-31 K {8..15,24..31}.
    const int koffB = (lane < 16) ? 0 : 8;

    // element-wise ownership: 2 (batch-row, h) pairs per thread; c in regs
    const int p0 = tid * 2;
    float c_st0 = 0.f, c_st1 = 0.f;

    __syncthreads();

    for (int t = 0; t < TT; ++t) {
        // ---- gates[16,512] = [x_t ; h] @ W^T + bias (bias in C init) -----
        v8f acc = {bn, bn, bn, bn, bn, bn, bn, bn};
        {
            const unsigned short* aX = Ax + (t & 1) * 16 * AXROW + arow * AXROW + koffA;
            const unsigned short* aH = Ah + arow * AHROW + koffA;
            const unsigned short* bB = Wl + ncol * WROW + koffB;
#pragma unroll
            for (int kc = 0; kc < XC; ++kc) {
                v16bf af = load_frag2(aX + kc * 32, aX + kc * 32 + 16);
                v16bf bf = load_frag2(bB + kc * 32, bB + kc * 32 + 16);
                acc = __builtin_amdgcn_wmma_f32_16x16x32_bf16(
                    false, af, false, bf, (short)0, acc, false, false);
            }
#pragma unroll
            for (int kc = 0; kc < HC; ++kc) {
                v16bf af = load_frag2(aH + kc * 32, aH + kc * 32 + 16);
                v16bf bf = load_frag2(bB + DIN + kc * 32, bB + DIN + kc * 32 + 16);
                acc = __builtin_amdgcn_wmma_f32_16x16x32_bf16(
                    false, af, false, bf, (short)0, acc, false, false);
            }
        }

        // ---- overlap: stage x_{t+1} into the other buffer ----------------
        if (t + 1 < TT) {
            unsigned short* dst = Ax + ((t + 1) & 1) * 16 * AXROW;
            for (int i = tid; i < 16 * DIN; i += 1024) {
                int r = i / DIN, d = i % DIN;
                dst[r * AXROW + d] =
                    to_bf_in(xin[(((size_t)(b0 + r)) * TT + (t + 1)) * DIN + d]);
            }
            if (tid < 16 && t + 2 < TT) {   // pull t+2 rows toward caches
                const TIN* nx = &xin[(((size_t)(b0 + tid)) * TT + (t + 2)) * DIN];
                __builtin_prefetch(nx, 0, 1);
                __builtin_prefetch(nx + DIN / 2, 0, 1);
            }
        }

        {   // D layout: lanes 0-15 N=lane rows 0..7; lanes 16-31 rows 8..15
            const int r0 = (lane < 16) ? 0 : 8;
#pragma unroll
            for (int r = 0; r < 8; ++r)
                Gl[(r0 + r) * GROW + ncol] = acc[r];
        }
        __syncthreads();

        // ---- LSTM element-wise (f32, hardware tanh) ----------------------
#pragma unroll
        for (int j = 0; j < 2; ++j) {
            const int p = p0 + j;
            const int r = p >> 7;       // batch row in tile
            const int h = p & (HH - 1);
            const float* g = &Gl[r * GROW];
            float ig = fast_sigmoid(g[h]);
            float fg = fast_sigmoid(g[HH + h]);
            float gg = fast_tanh(g[2 * HH + h]);
            float og = fast_sigmoid(g[3 * HH + h]);
            float c = fg * (j ? c_st1 : c_st0) + ig * gg;
            if (j) c_st1 = c; else c_st0 = c;
            float hv = og * fast_tanh(c);
            Ah[r * AHROW + h] = f2bf(hv);                           // GEMM input t+1
            store_h(&hout[(((size_t)(b0 + r)) * TT + t) * HH + h], hv);
        }
        __syncthreads();
    }
}

// ---------------------------------------------------------------------------
// Attention pooling over time + MLP head, f32, one workgroup per batch row.
// ---------------------------------------------------------------------------
__global__ __launch_bounds__(256)
void attn_mlp_kernel(const float* __restrict__ hs,      // [B][T][H]
                     const float* __restrict__ w_attn,  // [1][H]
                     const float* __restrict__ b_attn,  // [1]
                     const float* __restrict__ w1,      // [64][H]
                     const float* __restrict__ b1,      // [64]
                     const float* __restrict__ w2,      // [F][64]
                     const float* __restrict__ b2,      // [F]
                     float* __restrict__ out)           // [B][F]
{
    __shared__ float sc[TT];
    __shared__ float red[256];
    __shared__ float ctx[HH];
    __shared__ float h1s[64];
    const int b = blockIdx.x, tid = threadIdx.x;
    const float battn = b_attn[0];

    for (int t = tid; t < TT; t += 256) {
        const float* hp = hs + ((size_t)b * TT + t) * HH;
        float s = battn;
        for (int k = 0; k < HH; ++k) s += hp[k] * w_attn[k];
        sc[t] = s;
    }
    __syncthreads();

    float m = -1e30f;
    for (int t = tid; t < TT; t += 256) m = fmaxf(m, sc[t]);
    red[tid] = m; __syncthreads();
    for (int s = 128; s > 0; s >>= 1) {
        if (tid < s) red[tid] = fmaxf(red[tid], red[tid + s]);
        __syncthreads();
    }
    m = red[0];
    __syncthreads();

    float sum = 0.f;
    for (int t = tid; t < TT; t += 256) {
        float e = __expf(sc[t] - m);
        sc[t] = e;
        sum += e;
    }
    red[tid] = sum; __syncthreads();
    for (int s = 128; s > 0; s >>= 1) {
        if (tid < s) red[tid] += red[tid + s];
        __syncthreads();
    }
    const float inv = 1.f / red[0];

    if (tid < HH) {
        float a = 0.f;
        for (int t = 0; t < TT; ++t) a += sc[t] * hs[((size_t)b * TT + t) * HH + tid];
        ctx[tid] = a * inv;
    }
    __syncthreads();
    if (tid < 64) {
        float a = b1[tid];
        for (int k = 0; k < HH; ++k) a += ctx[k] * w1[tid * HH + k];
        h1s[tid] = fmaxf(a, 0.f);
    }
    __syncthreads();
    if (tid < FF) {
        float a = b2[tid];
        for (int k = 0; k < 64; ++k) a += h1s[k] * w2[tid * 64 + k];
        out[b * FF + tid] = a;
    }
}

// ---------------------------------------------------------------------------
extern "C" void kernel_launch(void* const* d_in, const int* in_sizes, int n_in,
                              void* d_out, int out_size, void* d_ws, size_t ws_size,
                              hipStream_t stream) {
    const float* x         = (const float*)d_in[0];
    const float* w_ih0     = (const float*)d_in[1];
    const float* w_ih_rest = (const float*)d_in[2];
    const float* w_hh      = (const float*)d_in[3];
    const float* b_ih      = (const float*)d_in[4];
    const float* b_hh      = (const float*)d_in[5];
    const float* w_attn    = (const float*)d_in[6];
    const float* b_attn    = (const float*)d_in[7];
    const float* w1        = (const float*)d_in[8];
    const float* b1        = (const float*)d_in[9];
    const float* w2        = (const float*)d_in[10];
    const float* b2        = (const float*)d_in[11];

    // Workspace layout (16B aligned):
    //   wp0   : bf16 [512][168]                 172,032 B  @ 0
    //   wp12  : bf16 [2][512][264]              540,672 B  @ 172,032
    //   biases: f32  [3][512]                     6,144 B  @ 712,704
    //   bufA  : 33,554,432 B @ 720,896  (bf16 hs1, later f32 hs3)
    //   bufB  : 16,777,216 B @ 34,275,328 (bf16 hs2)        total ~51 MB
    char* ws = (char*)d_ws;
    unsigned short* wp0    = (unsigned short*)ws;
    unsigned short* wp12   = (unsigned short*)(ws + 172032);
    float*          biases = (float*)(ws + 712704);
    char*           bufA   = ws + 720896;
    char*           bufB   = ws + 720896 + 33554432;

    pack_weights_kernel<<<512, 256, 0, stream>>>(w_ih0, w_ih_rest, w_hh,
                                                 b_ih, b_hh, wp0, wp12, biases);

    // Dynamic LDS: layer0  = 172,032 + 2,560 + 4,352 + 33,024 = 211,968 B
    //              layer1/2= 270,336 + 8,704 + 4,352 + 33,024 = 316,416 B (<320KB)
    lstm_layer_kernel<DD, float, unsigned short>
        <<<BB / 16, 1024, 211968, stream>>>(x, wp0, biases,
                                            (unsigned short*)bufA);
    lstm_layer_kernel<HH, unsigned short, unsigned short>
        <<<BB / 16, 1024, 316416, stream>>>((const unsigned short*)bufA,
                                            wp12, biases + GG,
                                            (unsigned short*)bufB);
    lstm_layer_kernel<HH, unsigned short, float>
        <<<BB / 16, 1024, 316416, stream>>>((const unsigned short*)bufB,
                                            wp12 + (size_t)GG * 264,
                                            biases + 2 * GG,
                                            (float*)bufA);

    attn_mlp_kernel<<<BB, 256, 0, stream>>>((const float*)bufA, w_attn, b_attn,
                                            w1, b1, w2, b2, (float*)d_out);
}
